// ImprovedSpikingWorldModel_55250459296346
// MI455X (gfx1250) — matrix operand order repaired and checked by reading, not compiled, not run on hardware
//
#include <hip/hip_runtime.h>
#include <hip/hip_bf16.h>
#include <cstdint>

// ---------------------------------------------------------------------------
// ImprovedSpikingWorldModel forward for MI455X (gfx1250, wave32, WMMA).
// All GEMMs go through v_wmma_f32_16x16x32_f16. A-tiles are staged with
// gfx1250 async direct-to-LDS loads (global_load_async_to_lds_b128 +
// s_wait_asynccnt); B-tiles via 128-bit loads + LDS transpose.
// h_input@w_in is hoisted out of the T=8 loop; w_rec runs as 3
// block-diagonal 256x256 WMMA GEMMs; per-step GEMMs initialize from X
// in the epilogue (no d2d copies).
// ---------------------------------------------------------------------------

typedef __attribute__((ext_vector_type(16))) _Float16 v16h;
typedef __attribute__((ext_vector_type(8)))  float    v8f;
typedef __attribute__((ext_vector_type(4)))  uint32_t v4u;
typedef _Float16 h16;

#define B_   8192
#define SD_  128
#define AD_  32
#define P_   8
#define H_   768
#define L_   2
#define T_   8
#define S_   3
#define F_   256
#define HEADN_ 1344   // 1024 dp | 128 dd | 128 lv | 64 rw1

// ============================ WMMA GEMM =====================================
// C[M,N](f32) = A[M,K](f16,row-major,lda) * B[K,N](f16,row-major,ldb)
// epilogue: +bias[col] (BIAS) | += C (ACC) | += Cinit (CIN, reads other buffer)
// Block: 256 threads = 8 waves. Tile 128(M) x 64(N), K-step 32.
// Wave grid 4x2; each wave computes a 32x32 tile = 2x2 WMMA fragments.
template <bool ACC, bool BIAS, bool CIN>
__global__ __launch_bounds__(256) void k_gemm(
    const h16* __restrict__ A, int lda,
    const h16* __restrict__ Bw, int ldb,
    float* __restrict__ C, int ldc,
    const float* __restrict__ bias,
    const float* __restrict__ Ci,
    int M, int N, int K)
{
  __shared__ __align__(16) h16 As[2][128][40];  // [m][k], 80B row stride (16B mult)
  __shared__ __align__(16) h16 Bs[2][64][40];   // [n][k] transposed

  const int tid  = threadIdx.x;
  const int lane = tid & 31;
  const int wave = tid >> 5;
  const int wm   = (wave & 3) * 32;   // wave row offset in tile
  const int wn   = (wave >> 2) * 32;  // wave col offset in tile
  const int m0   = blockIdx.y * 128;
  const int n0   = blockIdx.x * 64;
  const int nkt  = K >> 5;

  v8f acc[2][2];
  const v8f vz = {0.f, 0.f, 0.f, 0.f, 0.f, 0.f, 0.f, 0.f};
  acc[0][0] = vz; acc[0][1] = vz; acc[1][0] = vz; acc[1][1] = vz;

  auto loadTiles = [&](int kt, int buf) {
    const int k0 = kt << 5;
    // ---- A tile 128x32 halves = 512 x 16B chunks: async DMA straight to LDS
#pragma unroll
    for (int i = 0; i < 2; ++i) {
      int c = tid + i * 256;          // 0..511
      int r = c >> 2, q = c & 3;      // row, 16B-chunk within row
      const h16* gp = &A[(size_t)(m0 + r) * lda + k0 + q * 8];
      uint32_t lp = (uint32_t)(uintptr_t)(const void*)&As[buf][r][q * 8];
      asm volatile("global_load_async_to_lds_b128 %0, %1, off"
                   :: "v"(lp), "v"(gp) : "memory");
    }
    // ---- B tile 32x64 halves = 256 x 16B chunks (1/thread): b128 load + LDS transpose
    {
      int k  = tid >> 3;              // 0..31
      int n8 = (tid & 7) * 8;         // 0..56
      v4u u = *reinterpret_cast<const v4u*>(&Bw[(size_t)(k0 + k) * ldb + n0 + n8]);
      union { uint32_t u32; h16 h[2]; } cv;
#pragma unroll
      for (int j = 0; j < 4; ++j) {
        cv.u32 = u[j];
        Bs[buf][n8 + 2 * j + 0][k] = cv.h[0];
        Bs[buf][n8 + 2 * j + 1][k] = cv.h[1];
      }
    }
    // keep L2 warm for the tile after next (global_prefetch_b8)
    if (kt + 2 < nkt) {
      __builtin_prefetch(&A[(size_t)(m0 + (tid & 127)) * lda + (size_t)(kt + 2) * 32], 0, 1);
      __builtin_prefetch(&Bw[(size_t)((kt + 2) * 32 + (tid & 31)) * ldb + n0], 0, 1);
    }
  };

  // A fragment (16x32 f16): lane L holds row M=L%16; VGPR v<4 -> K=(L<16?0:8)+2v,
  // v>=4 -> K=16+(L<16?0:8)+2(v-4).
  auto fragA = [&](int buf, int mb) -> v16h {
    v16h f;
    const int r  = mb + (lane & 15);
    const int kh = (lane < 16) ? 0 : 8;
#pragma unroll
    for (int v = 0; v < 8; ++v) {
      int k = (v < 4) ? (kh + 2 * v) : (16 + kh + 2 * (v - 4));
      f[2 * v]     = As[buf][r][k];
      f[2 * v + 1] = As[buf][r][k + 1];
    }
    return f;
  };
  // B fragment (32x16 f16): lane L holds col N=L%16; same K order as A fragment.
  auto fragB = [&](int buf, int nb) -> v16h {
    v16h f;
    const int n  = nb + (lane & 15);
    const int kb = (lane < 16) ? 0 : 8;
#pragma unroll
    for (int v = 0; v < 8; ++v) {
      int k = (v < 4) ? (kb + 2 * v) : (16 + kb + 2 * (v - 4));
      f[2 * v]     = Bs[buf][n][k];
      f[2 * v + 1] = Bs[buf][n][k + 1];
    }
    return f;
  };

  loadTiles(0, 0);
  asm volatile("s_wait_asynccnt 0x0" ::: "memory");
  __syncthreads();
  for (int kt = 0; kt < nkt; ++kt) {
    const int buf = kt & 1;
    if (kt + 1 < nkt) loadTiles(kt + 1, buf ^ 1);
    v16h a0 = fragA(buf, wm), a1 = fragA(buf, wm + 16);
    v16h b0 = fragB(buf, wn), b1 = fragB(buf, wn + 16);
    acc[0][0] = __builtin_amdgcn_wmma_f32_16x16x32_f16(false, a0, false, b0, (short)0, acc[0][0], false, false);
    acc[0][1] = __builtin_amdgcn_wmma_f32_16x16x32_f16(false, a0, false, b1, (short)0, acc[0][1], false, false);
    acc[1][0] = __builtin_amdgcn_wmma_f32_16x16x32_f16(false, a1, false, b0, (short)0, acc[1][0], false, false);
    acc[1][1] = __builtin_amdgcn_wmma_f32_16x16x32_f16(false, a1, false, b1, (short)0, acc[1][1], false, false);
    asm volatile("s_wait_asynccnt 0x0" ::: "memory");
    __syncthreads();
  }

  // C/D layout: VGPR p -> row = (lane<16 ? p : p+8), col = lane%16.
#pragma unroll
  for (int i = 0; i < 2; ++i) {
#pragma unroll
    for (int j = 0; j < 2; ++j) {
      const int cm  = m0 + wm + i * 16;
      const int cn  = n0 + wn + j * 16;
      const int col = cn + (lane & 15);
      const int rb  = cm + ((lane < 16) ? 0 : 8);
#pragma unroll
      for (int p = 0; p < 8; ++p) {
        size_t off = (size_t)(rb + p) * ldc + col;
        float val = acc[i][j][p];
        if (BIAS) val += bias[col];
        if (CIN)  val += Ci[off];
        if (ACC)  val += C[off];
        C[off] = val;
      }
    }
  }
}

// ============================ elementwise kernels ===========================

__global__ __launch_bounds__(256) void k_cast_f2h(const float* __restrict__ s,
                                                  h16* __restrict__ d, int n) {
  int i = blockIdx.x * 256 + threadIdx.x;
  if (i < n) d[i] = (h16)s[i];
}

// w_in[l] : [S,H,F] -> dst [H, S*F]  (dst[h*768 + s*256+f] = src[s*H*F + h*F + f])
__global__ __launch_bounds__(256) void k_cast_win(const float* __restrict__ src,
                                                  h16* __restrict__ dst) {
  int i = blockIdx.x * 256 + threadIdx.x;
  if (i >= H_ * H_) return;
  int h = i / H_, c = i % H_;
  int s = c >> 8, f = c & 255;
  dst[i] = (h16)src[(size_t)s * H_ * F_ + (size_t)h * F_ + f];
}

// Fused head weight [768 x 1344] and bias [1344]
__global__ __launch_bounds__(256) void k_build_head(
    const float* __restrict__ dpw, const float* __restrict__ ddw,
    const float* __restrict__ lvw, const float* __restrict__ rw1,
    const float* __restrict__ dpb, const float* __restrict__ ddb,
    const float* __restrict__ lvb, const float* __restrict__ rw1b,
    h16* __restrict__ W, float* __restrict__ bias)
{
  int i = blockIdx.x * 256 + threadIdx.x;
  if (i >= H_ * HEADN_) return;
  int k = i / HEADN_, c = i % HEADN_;
  float v, b;
  if (c < 1024)      { v = dpw[(size_t)k * 1024 + c];         b = dpb[c]; }
  else if (c < 1152) { v = ddw[(size_t)k * 128 + (c - 1024)]; b = ddb[c - 1024]; }
  else if (c < 1280) { v = lvw[(size_t)k * 128 + (c - 1152)]; b = lvb[c - 1152]; }
  else               { v = rw1[(size_t)k * 64  + (c - 1280)]; b = rw1b[c - 1280]; }
  W[i] = (h16)v;
  if (k == 0) bias[c] = b;
}

// Gaussian population encoder -> f16 enc[B, 1280]
__global__ __launch_bounds__(256) void k_encode(
    const float* __restrict__ state, const float* __restrict__ action,
    const float* __restrict__ ls_s, const float* __restrict__ ls_a,
    h16* __restrict__ enc)
{
  int i = blockIdx.x * 256 + threadIdx.x;   // exact grid B*1280
  int b = i / ((SD_ + AD_) * P_);
  int c = i % ((SD_ + AD_) * P_);
  float x, sig; int p;
  if (c < SD_ * P_) { int d = c >> 3; p = c & 7; x = state[(size_t)b * SD_ + d];  sig = expf(ls_s[p]); }
  else { int c2 = c - SD_ * P_; int d = c2 >> 3; p = c2 & 7; x = action[(size_t)b * AD_ + d]; sig = expf(ls_a[p]); }
  float cen = -1.0f + (2.0f / 7.0f) * (float)p;
  float t = (x - cen) / sig;
  enc[i] = (h16)expf(-0.5f * t * t);
}

// LayerNorm over 768 cols, optional fused SiLU; writes f32 (if Y) and f16 (if Yh)
template <bool SILU>
__global__ __launch_bounds__(256) void k_layernorm(
    const float* __restrict__ X, const float* __restrict__ g,
    const float* __restrict__ b, float* __restrict__ Y, h16* __restrict__ Yh)
{
  __shared__ float red[256];
  const int row = blockIdx.x, tid = threadIdx.x;
  const float* x = X + (size_t)row * H_;
  float v0 = x[tid], v1 = x[tid + 256], v2 = x[tid + 512];
  red[tid] = v0 + v1 + v2;
  __syncthreads();
  for (int o = 128; o > 0; o >>= 1) { if (tid < o) red[tid] += red[tid + o]; __syncthreads(); }
  float mean = red[0] * (1.0f / H_);
  __syncthreads();
  float d0 = v0 - mean, d1 = v1 - mean, d2 = v2 - mean;
  red[tid] = d0 * d0 + d1 * d1 + d2 * d2;
  __syncthreads();
  for (int o = 128; o > 0; o >>= 1) { if (tid < o) red[tid] += red[tid + o]; __syncthreads(); }
  float rstd = rsqrtf(red[0] * (1.0f / H_) + 1e-5f);
  float dd[3] = {d0, d1, d2};
#pragma unroll
  for (int j = 0; j < 3; ++j) {
    int c = tid + j * 256;
    float y = dd[j] * rstd * g[c] + b[c];
    if (SILU) y = y / (1.0f + expf(-y));
    if (Y)  Y[(size_t)row * H_ + c] = y;
    if (Yh) Yh[(size_t)row * H_ + c] = (h16)y;
  }
}

// LIF layer 0: v/adapt update + hard spikes (forward of straight-through _spike)
__global__ __launch_bounds__(256) void k_lif0(
    const float* __restrict__ cur, float* __restrict__ v, float* __restrict__ a,
    const float* __restrict__ ltm, const float* __restrict__ lta,
    h16* __restrict__ spkh)
{
  int i = blockIdx.x * 256 + threadIdx.x;   // exact grid B*H
  int h = i % H_;
  float alpha = expf(-expf(-ltm[h]));       // exp(-1/tau), tau = exp(ltm)
  float rho   = expf(-expf(-lta[h]));
  float c = cur[i], vv = v[i], aa = a[i];
  float vm  = alpha * vv + (1.0f - alpha) * c;
  float spk = (vm - (1.0f + 0.5f * aa)) > 0.0f ? 1.0f : 0.0f;
  v[i] = vm * (1.0f - spk);
  a[i] = rho * aa + spk;
  spkh[i] = (h16)spk;
}

// LIF layer 1 + output accumulators (s_acc over last-layer residual spike path)
__global__ __launch_bounds__(256) void k_lif1(
    const float* __restrict__ cur, float* __restrict__ v, float* __restrict__ a,
    const float* __restrict__ ltm, const float* __restrict__ lta,
    h16* __restrict__ spkh, const h16* __restrict__ spk0h,
    const float* __restrict__ hinp,
    float* __restrict__ s_acc, float* __restrict__ m_acc)
{
  int i = blockIdx.x * 256 + threadIdx.x;
  int h = i % H_;
  float alpha = expf(-expf(-ltm[h]));
  float rho   = expf(-expf(-lta[h]));
  float c = cur[i], vv = v[i], aa = a[i];
  float vm  = alpha * vv + (1.0f - alpha) * c;
  float spk = (vm - (1.0f + 0.5f * aa)) > 0.0f ? 1.0f : 0.0f;
  v[i] = vm * (1.0f - spk);
  a[i] = rho * aa + spk;
  spkh[i] = (h16)spk;
  s_acc[i] += spk + (float)spk0h[i] + hinp[i];  // spk1 + (spk0 + h_input residual)
  m_acc[i] += vm;
}

// combined = [s_acc/T , m_acc/T] as f16 [B, 1536]
__global__ __launch_bounds__(256) void k_combine(
    const float* __restrict__ s_acc, const float* __restrict__ m_acc,
    h16* __restrict__ comb)
{
  int i = blockIdx.x * 256 + threadIdx.x;   // exact grid B*1536
  int b = i / (2 * H_), c = i % (2 * H_);
  float v = (c < H_) ? s_acc[(size_t)b * H_ + c] : m_acc[(size_t)b * H_ + (c - H_)];
  comb[i] = (h16)(v * (1.0f / (float)T_));
}

// heads: softmax pop-decode + direct delta, logvar clip, reward reduce
__global__ __launch_bounds__(128) void k_heads(
    const float* __restrict__ ho, const float* __restrict__ state,
    const float* __restrict__ rw_w2, const float* __restrict__ rw_b2,
    float* __restrict__ out)
{
  __shared__ float red[64];
  const int b = blockIdx.x, t = threadIdx.x;   // t in [0,128)
  const float* h = ho + (size_t)b * HEADN_;
  float mx = -1e30f, lg[P_];
#pragma unroll
  for (int p = 0; p < P_; ++p) { lg[p] = h[t * P_ + p]; mx = fmaxf(mx, lg[p]); }
  float se = 0.f, dec = 0.f;
#pragma unroll
  for (int p = 0; p < P_; ++p) {
    float e = expf(lg[p] - mx);
    se += e;
    dec += e * (-1.0f + (2.0f / 7.0f) * (float)p);
  }
  dec /= se;
  float dir = h[1024 + t];
  out[(size_t)b * SD_ + t] = state[(size_t)b * SD_ + t] + 0.7f * dec + 0.3f * dir;
  float lv = h[1152 + t];
  out[(size_t)(B_ * SD_ + B_) + (size_t)b * SD_ + t] = fminf(fmaxf(lv, -10.0f), 2.0f);
  if (t < 64) {
    float z = h[1280 + t];
    z = z / (1.0f + expf(-z));
    red[t] = z * rw_w2[t];
  }
  __syncthreads();
  if (t < 32) red[t] += red[t + 32];
  __syncthreads();
  if (t == 0) {
    float acc = 0.f;
    for (int j = 0; j < 32; ++j) acc += red[j];
    out[(size_t)B_ * SD_ + b] = acc + rw_b2[0];
  }
}

// ================================ host ======================================

extern "C" void kernel_launch(void* const* d_in, const int* in_sizes, int n_in,
                              void* d_out, int out_size, void* d_ws, size_t ws_size,
                              hipStream_t stream)
{
  (void)in_sizes; (void)out_size; (void)ws_size;
  if (n_in < 31) return;

  // setup_inputs() dict order: state, action, then params in insertion order
  const float* state   = (const float*)d_in[0];
  const float* action  = (const float*)d_in[1];
  const float* ls_s    = (const float*)d_in[2];
  const float* ls_a    = (const float*)d_in[3];
  const float* fc_in_w = (const float*)d_in[4];
  const float* fc_in_b = (const float*)d_in[5];
  const float* ln_in_g = (const float*)d_in[6];
  const float* ln_in_b = (const float*)d_in[7];
  const float* w_in    = (const float*)d_in[8];   // [2,3,768,256]
  const float* b_in    = (const float*)d_in[9];   // [2,3,256] -> layer slice 768 contiguous
  const float* w_rec   = (const float*)d_in[10];  // [2,3,256,256]
  const float* ltm     = (const float*)d_in[11];  // [2,3,256]
  const float* lta     = (const float*)d_in[12];
  const float* ro_w1   = (const float*)d_in[13];
  const float* ro_b1   = (const float*)d_in[14];
  const float* ro_g1   = (const float*)d_in[15];
  const float* ro_bn1  = (const float*)d_in[16];
  const float* ro_w2   = (const float*)d_in[17];
  const float* ro_b2   = (const float*)d_in[18];
  const float* ro_g2   = (const float*)d_in[19];
  const float* ro_bn2  = (const float*)d_in[20];
  const float* dp_w    = (const float*)d_in[21];
  const float* dp_b    = (const float*)d_in[22];
  const float* dd_w    = (const float*)d_in[23];
  const float* dd_b    = (const float*)d_in[24];
  const float* rw_w1   = (const float*)d_in[25];
  const float* rw_b1   = (const float*)d_in[26];
  const float* rw_w2   = (const float*)d_in[27];
  const float* rw_b2   = (const float*)d_in[28];
  const float* lv_w    = (const float*)d_in[29];
  const float* lv_b    = (const float*)d_in[30];

  uint8_t* ws = (uint8_t*)d_ws;
  size_t off = 0;
  auto alloc = [&](size_t bytes) -> void* {
    void* p = ws + off;
    off = (off + bytes + 255) & ~(size_t)255;
    return p;
  };
  const size_t SZf = (size_t)B_ * H_ * sizeof(float);
  const size_t SZh = (size_t)B_ * H_ * sizeof(h16);

  h16*   Wfc    = (h16*)  alloc((size_t)1280 * H_ * 2);
  h16*   Win0   = (h16*)  alloc((size_t)H_ * H_ * 2);
  h16*   Win1   = (h16*)  alloc((size_t)H_ * H_ * 2);
  h16*   Wrec0  = (h16*)  alloc((size_t)S_ * F_ * F_ * 2);
  h16*   Wrec1  = (h16*)  alloc((size_t)S_ * F_ * F_ * 2);
  h16*   Wro1   = (h16*)  alloc((size_t)2 * H_ * H_ * 2);
  h16*   Wro2   = (h16*)  alloc((size_t)H_ * H_ * 2);
  h16*   Whead  = (h16*)  alloc((size_t)H_ * HEADN_ * 2);
  float* bhead  = (float*)alloc((size_t)HEADN_ * 4);
  h16*   enc16  = (h16*)  alloc((size_t)B_ * 1280 * 2);
  float* tmp1   = (float*)alloc(SZf);                 // pre-LN GEMM outputs
  float* hinp   = (float*)alloc(SZf);
  h16*   hinp16 = (h16*)  alloc(SZh);
  float* X0     = (float*)alloc(SZf);
  float* X1     = (float*)alloc(SZf);
  float* v0     = (float*)alloc(SZf);
  float* a0     = (float*)alloc(SZf);
  float* v1     = (float*)alloc(SZf);
  float* a1     = (float*)alloc(SZf);
  h16*   spk0h  = (h16*)  alloc(SZh);
  h16*   spk1h  = (h16*)  alloc(SZh);
  float* cur    = (float*)alloc(SZf);
  float* s_acc  = (float*)alloc(SZf);
  float* m_acc  = (float*)alloc(SZf);
  h16*   comb16 = (h16*)  alloc((size_t)B_ * 2 * H_ * 2);
  h16*   h1_16  = (h16*)  alloc(SZh);
  h16*   hid16  = (h16*)  alloc(SZh);
  float* headout= (float*)alloc((size_t)B_ * HEADN_ * 4);

  // mode: 0 = store(+bias), 1 = accumulate into C, 2 = initialize from Ci
  auto gemm = [&](const h16* A, int lda, const h16* Bw, int ldb, float* C, int ldc,
                  const float* bias, const float* Ci, int M, int N, int K, int mode) {
    dim3 g(N / 64, M / 128);
    if (mode == 1)
      k_gemm<true, false, false><<<g, dim3(256), 0, stream>>>(A, lda, Bw, ldb, C, ldc, nullptr, nullptr, M, N, K);
    else if (mode == 2)
      k_gemm<false, false, true><<<g, dim3(256), 0, stream>>>(A, lda, Bw, ldb, C, ldc, nullptr, Ci, M, N, K);
    else
      k_gemm<false, true, false><<<g, dim3(256), 0, stream>>>(A, lda, Bw, ldb, C, ldc, bias, nullptr, M, N, K);
  };
  auto cast = [&](const float* s, h16* d, int n) {
    k_cast_f2h<<<(n + 255) / 256, 256, 0, stream>>>(s, d, n);
  };

  // ---- 0) weight preparation (f16 copies / re-layouts) ----
  cast(fc_in_w, Wfc, 1280 * H_);
  k_cast_win<<<(H_ * H_ + 255) / 256, 256, 0, stream>>>(w_in, Win0);
  k_cast_win<<<(H_ * H_ + 255) / 256, 256, 0, stream>>>(w_in + (size_t)S_ * H_ * F_, Win1);
  cast(w_rec, Wrec0, S_ * F_ * F_);
  cast(w_rec + (size_t)S_ * F_ * F_, Wrec1, S_ * F_ * F_);
  cast(ro_w1, Wro1, 2 * H_ * H_);
  cast(ro_w2, Wro2, H_ * H_);
  k_build_head<<<(H_ * HEADN_ + 255) / 256, 256, 0, stream>>>(
      dp_w, dd_w, lv_w, rw_w1, dp_b, dd_b, lv_b, rw_b1, Whead, bhead);

  // ---- 1) population encode + fc_in + LN ----
  k_encode<<<(B_ * 1280) / 256, 256, 0, stream>>>(state, action, ls_s, ls_a, enc16);
  gemm(enc16, 1280, Wfc, H_, tmp1, H_, fc_in_b, nullptr, B_, H_, 1280, 0);
  k_layernorm<false><<<B_, 256, 0, stream>>>(tmp1, ln_in_g, ln_in_b, hinp, hinp16);

  // ---- 2) time-invariant input projections X_l = h_input @ w_in[l] + b_in[l] ----
  gemm(hinp16, H_, Win0, H_, X0, H_, b_in,      nullptr, B_, H_, H_, 0);
  gemm(hinp16, H_, Win1, H_, X1, H_, b_in + H_, nullptr, B_, H_, H_, 0);

  // ---- 3) zero recurrent state + accumulators ----
  hipMemsetAsync(v0, 0, SZf, stream);  hipMemsetAsync(a0, 0, SZf, stream);
  hipMemsetAsync(v1, 0, SZf, stream);  hipMemsetAsync(a1, 0, SZf, stream);
  hipMemsetAsync(spk0h, 0, SZh, stream); hipMemsetAsync(spk1h, 0, SZh, stream);
  hipMemsetAsync(s_acc, 0, SZf, stream); hipMemsetAsync(m_acc, 0, SZf, stream);

  // ---- 4) T=8 spiking steps ----
  const int ew_grid = (B_ * H_) / 256;
  for (int t = 0; t < T_; ++t) {
    // layer 0: cur = X0 + prev0 @ w_rec[0]  (block-diagonal, 3 disjoint col slices)
    for (int s = 0; s < S_; ++s)
      gemm(spk0h + s * F_, H_, Wrec0 + (size_t)s * F_ * F_, F_,
           cur + s * F_, H_, nullptr, X0 + s * F_, B_, F_, F_, 2);
    k_lif0<<<ew_grid, 256, 0, stream>>>(cur, v0, a0, ltm, lta, spk0h);

    // layer 1: cur = X1 + spk0 @ w_in[1] + prev1 @ w_rec[1]
    gemm(spk0h, H_, Win1, H_, cur, H_, nullptr, X1, B_, H_, H_, 2);
    for (int s = 0; s < S_; ++s)
      gemm(spk1h + s * F_, H_, Wrec1 + (size_t)s * F_ * F_, F_,
           cur + s * F_, H_, nullptr, nullptr, B_, F_, F_, 1);
    k_lif1<<<ew_grid, 256, 0, stream>>>(cur, v1, a1, ltm + H_, lta + H_,
                                        spk1h, spk0h, hinp, s_acc, m_acc);
  }

  // ---- 5) readout MLP ----
  k_combine<<<(B_ * 2 * H_) / 256, 256, 0, stream>>>(s_acc, m_acc, comb16);
  gemm(comb16, 2 * H_, Wro1, H_, tmp1, H_, ro_b1, nullptr, B_, H_, 2 * H_, 0);
  k_layernorm<true><<<B_, 256, 0, stream>>>(tmp1, ro_g1, ro_bn1, nullptr, h1_16);
  gemm(h1_16, H_, Wro2, H_, tmp1, H_, ro_b2, nullptr, B_, H_, H_, 0);
  k_layernorm<false><<<B_, 256, 0, stream>>>(tmp1, ro_g2, ro_bn2, nullptr, hid16);

  // ---- 6) fused heads GEMM + decode ----
  gemm(hid16, H_, Whead, HEADN_, headout, HEADN_, bhead, nullptr, B_, HEADN_, H_, 0);
  k_heads<<<B_, 128, 0, stream>>>(headout, state, rw_w2, rw_b2, (float*)d_out);
}
